// SelfAttentionBlock_26517128085934
// MI455X (gfx1250) — matrix operand order repaired and checked by reading, not compiled, not run on hardware
//
#include <hip/hip_runtime.h>
#include <math.h>

typedef __attribute__((ext_vector_type(16))) __bf16 v16bf;
typedef __attribute__((ext_vector_type(8)))  float  v8f;
typedef __attribute__((ext_vector_type(4)))  unsigned int v4u;
typedef __attribute__((ext_vector_type(8)))  unsigned int v8u;

constexpr int BB = 8;      // batch
constexpr int NN = 1024;   // sequence
constexpr int DD = 2048;   // model dim
constexpr int HH = 16;     // heads
constexpr int dh = 128;    // head dim

// --- WMMA bf16 fragment index helpers (CDNA5 ISA layouts) ---
__device__ __forceinline__ int a_kidx(int v, int lane) {
  int kb = (lane & 16) ? 8 : 0;
  return (v < 4) ? (kb + 2 * v) : (16 + kb + 2 * (v - 4));
}
__device__ __forceinline__ int b_kidx(int v, int lane) {
  return a_kidx(v, lane);
}

__device__ __forceinline__ v8f wmma_bf16(const v16bf& a, const v16bf& b, const v8f& c) {
  return __builtin_amdgcn_wmma_f32_16x16x32_bf16(false, a, false, b, (short)0, c, false, false);
}

union Frag { v16bf v; unsigned int u[8]; };
union H8  { uint4 u; __bf16 h[8]; };
union H4  { unsigned long long u; __bf16 h[4]; };

// CDNA5 async copy: global -> LDS, 16 bytes per lane, tracked by ASYNCcnt.
__device__ __forceinline__ void async_copy_b128(unsigned int lds_off,
                                                const __bf16* gaddr) {
  asm volatile("global_load_async_to_lds_b128 %0, %1, off"
               :: "v"(lds_off), "v"((unsigned long long)(size_t)gaddr)
               : "memory");
}
__device__ __forceinline__ void wait_async0() {
  asm volatile("s_wait_asynccnt 0" ::: "memory");
}
__device__ __forceinline__ unsigned int lds_off32(const void* p) {
  // flat LDS aperture keeps the LDS byte offset in addr[31:0]
  return (unsigned int)(size_t)p;
}

// ----------------------------------------------------------------------
// Tensor Data Mover: 2D tile load global->LDS (CDNA5 ISA).
// tile = tile_rows x 32 bf16 elements, tensor row stride = 2048 elements,
// LDS padding: 4 DWORDs after every 16 DWORDs -> 80B padded rows.
// ----------------------------------------------------------------------
__device__ __forceinline__ void tdm_load_tile(unsigned int lds_off,
                                              const __bf16* gp,
                                              unsigned int tile_rows) {
  unsigned long long ga = (unsigned long long)(size_t)gp;
  v4u g0;
  g0[0] = 1u;                                   // count=1, no gather
  g0[1] = lds_off;                              // lds_addr
  g0[2] = (unsigned int)ga;                     // global_addr[31:0]
  g0[3] = (unsigned int)((ga >> 32) & 0x01FFFFFFu) | 0x80000000u; // addr[56:32] | type=2
  v8u g1;
  g1[0] = (1u << 16)        // data_size = 2 bytes
        | (1u << 20)        // pad_enable
        | (3u << 22)        // pad_interval: 16 DWORDs (64B = one tile row)
        | (3u << 25);       // pad_amount: 4 DWORDs (16B)
  g1[1] = (2048u & 0xFFFFu) << 16;              // tensor_dim0 = 2048 (lo16)
  g1[2] = 0xFFFF0000u;                          // dim0 hi | tensor_dim1 = 0xFFFF
  g1[3] = (32u << 16);                          // dim1 hi | tile_dim0 = 32
  g1[4] = tile_rows & 0xFFFFu;                  // tile_dim1 | tile_dim2 = 0
  g1[5] = 2048u;                                // tensor_dim0_stride lo32
  g1[6] = 0u;
  g1[7] = 0u;
  asm volatile("tensor_load_to_lds %0, %1" :: "s"(g0), "s"(g1) : "memory");
}

// ======================================================================
// Kernel 0a: elementwise fp32 -> bf16 (x)
// ======================================================================
__global__ __launch_bounds__(256)
void convert_bf16_kernel(const float* __restrict__ in, __bf16* __restrict__ out,
                         int n4) {
  int i = blockIdx.x * blockDim.x + threadIdx.x;
  int stride = gridDim.x * blockDim.x;
  const float4* in4 = reinterpret_cast<const float4*>(in);
  unsigned long long* out4 = reinterpret_cast<unsigned long long*>(out);
  for (; i < n4; i += stride) {
    float4 f = in4[i];
    H4 o;
    o.h[0] = (__bf16)f.x; o.h[1] = (__bf16)f.y;
    o.h[2] = (__bf16)f.z; o.h[3] = (__bf16)f.w;
    out4[i] = o.u;
  }
}

// ======================================================================
// Kernel 0b: W [k][n] fp32 -> Wt [n][k] bf16, LDS-tiled 64x64
// ======================================================================
__global__ __launch_bounds__(256)
void transpose_bf16_kernel(const float* __restrict__ W, __bf16* __restrict__ Wt) {
  __shared__ __bf16 t[64][72];
  const int tid = threadIdx.x;
  const int k0 = blockIdx.y * 64;
  const int n0 = blockIdx.x * 64;
  for (int i = 0; i < 4; ++i) {
    int idx = tid + i * 256;
    int r = idx >> 4, c = idx & 15;
    float4 f = *reinterpret_cast<const float4*>(W + (size_t)(k0 + r) * DD + n0 + c * 4);
    t[r][c * 4 + 0] = (__bf16)f.x; t[r][c * 4 + 1] = (__bf16)f.y;
    t[r][c * 4 + 2] = (__bf16)f.z; t[r][c * 4 + 3] = (__bf16)f.w;
  }
  __syncthreads();
  for (int i = 0; i < 16; ++i) {
    int idx = tid + i * 256;
    int nr = idx >> 6, kc = idx & 63;
    Wt[(size_t)(n0 + nr) * DD + k0 + kc] = t[kc][nr];
  }
}

// ======================================================================
// Kernel 1: fused QKV projection  out = bf16(xb @ Wt^T + bias)
//   TDM-staged, double-buffered; 128x128 tile, K-step 32; 8 waves x 16x128
// ======================================================================
__global__ __launch_bounds__(256)
void qkv_gemm_kernel(const __bf16* __restrict__ xb,
                     const __bf16* __restrict__ Wqt, const float* __restrict__ bq,
                     const __bf16* __restrict__ Wkt, const float* __restrict__ bk,
                     const __bf16* __restrict__ Wvt, const float* __restrict__ bv,
                     __bf16* __restrict__ Qb, __bf16* __restrict__ Kb,
                     __bf16* __restrict__ Vb)
{
  __shared__ __bf16 As[2][128][40];   // double-buffered, 80B padded rows
  __shared__ __bf16 Bt[2][128][40];

  const __bf16* Wt; const float* bias; __bf16* out;
  if (blockIdx.z == 0)      { Wt = Wqt; bias = bq; out = Qb; }
  else if (blockIdx.z == 1) { Wt = Wkt; bias = bk; out = Kb; }
  else                      { Wt = Wvt; bias = bv; out = Vb; }

  const int tid  = threadIdx.x;
  const int lane = tid & 31;
  const int wv   = tid >> 5;
  const int row0 = blockIdx.y * 128;
  const int col0 = blockIdx.x * 128;

  const __bf16* aGlob = xb + (size_t)row0 * DD;   // +kt*32 per step
  const __bf16* bGlob = Wt + (size_t)col0 * DD;

  v8f acc[8];
  const v8f vzero = {};
  for (int i = 0; i < 8; ++i) acc[i] = vzero;

  constexpr int NT = DD / 32;   // 64 K-steps

  // prologue: TDM-stage tile 0 into buffer 0 (wave 0 drives the DMA)
  if (wv == 0) {
    tdm_load_tile(lds_off32(&As[0][0][0]), aGlob, 128);
    tdm_load_tile(lds_off32(&Bt[0][0][0]), bGlob, 128);
  }

  for (int kt = 0; kt < NT; ++kt) {
    const int cur = kt & 1;
    if (wv == 0) {
      if (kt + 1 < NT) {
        // prefetch next tile into the other buffer, then wait for current
        tdm_load_tile(lds_off32(&As[cur ^ 1][0][0]), aGlob + (kt + 1) * 32, 128);
        tdm_load_tile(lds_off32(&Bt[cur ^ 1][0][0]), bGlob + (kt + 1) * 32, 128);
        __builtin_amdgcn_s_wait_tensorcnt(2);
      } else {
        __builtin_amdgcn_s_wait_tensorcnt(0);
      }
    }
    __syncthreads();

    Frag af;
    const int arow = wv * 16 + (lane & 15);
    for (int v = 0; v < 8; ++v)
      af.u[v] = *reinterpret_cast<const unsigned int*>(&As[cur][arow][a_kidx(v, lane)]);

    for (int nt = 0; nt < 8; ++nt) {
      Frag bfr;
      const int bcol = nt * 16 + (lane & 15);
      for (int v = 0; v < 8; ++v)
        bfr.u[v] = *reinterpret_cast<const unsigned int*>(&Bt[cur][bcol][b_kidx(v, lane)]);
      acc[nt] = wmma_bf16(af.v, bfr.v, acc[nt]);
    }
    __syncthreads();
  }

  const int rbase = wv * 16 + ((lane & 16) ? 8 : 0);
  const int cl    = (lane & 15);
  for (int nt = 0; nt < 8; ++nt) {
    int gcol = col0 + nt * 16 + cl;
    float bsc = bias[gcol];
    for (int r = 0; r < 8; ++r) {
      int grow = row0 + rbase + r;
      out[(size_t)grow * DD + gcol] = (__bf16)(acc[nt][r] + bsc);
    }
  }
}

// ======================================================================
// Kernel 2: flash attention + residual
//   K staged via async copy; V staged transposed; P via per-wave LDS
// ======================================================================
__global__ __launch_bounds__(256)
void flash_attn_kernel(const float* __restrict__ x,
                       const __bf16* __restrict__ Qb,
                       const __bf16* __restrict__ Kb,
                       const __bf16* __restrict__ Vb,
                       float* __restrict__ outp)
{
  __shared__ __bf16 Ks[32][136];     // 32 keys x 128 d (272B rows)
  __shared__ __bf16 Vt[128][40];     // 128 d x 32 keys (transposed)
  __shared__ __bf16 Pb[8][16][36];   // per-wave P buffer 16x32

  const int tid  = threadIdx.x;
  const int lane = tid & 31;
  const int wv   = tid >> 5;
  const int b    = blockIdx.x / HH;
  const int h    = blockIdx.x % HH;
  const int q0   = blockIdx.y * 128 + wv * 16;

  Frag aq[4];
  {
    const __bf16* qp = Qb + (size_t)(b * NN + q0 + (lane & 15)) * DD + h * dh;
    for (int c = 0; c < 4; ++c)
      for (int v = 0; v < 8; ++v)
        aq[c].u[v] = *reinterpret_cast<const unsigned int*>(qp + c * 32 + a_kidx(v, lane));
  }

  const v8f vzero = {};
  v8f acc[8];
  for (int i = 0; i < 8; ++i) acc[i] = vzero;
  float mrow[8], lrow[8];
  for (int r = 0; r < 8; ++r) { mrow[r] = -1e30f; lrow[r] = 0.0f; }

  const float scale = 0.08838834764831845f;  // 1/sqrt(128)

  for (int mt = 0; mt < NN / 32; ++mt) {
    for (int i = 0; i < 2; ++i) {
      int idx = tid + i * 256;
      int r = idx >> 4, c = idx & 15;
      async_copy_b128(lds_off32(&Ks[r][c * 8]),
                      Kb + (size_t)(b * NN + mt * 32 + r) * DD + h * dh + c * 8);
    }
    for (int i = 0; i < 2; ++i) {
      int idx = tid + i * 256;
      int r = idx >> 4, c = idx & 15;
      H8 tv;
      tv.u = *reinterpret_cast<const uint4*>(
          Vb + (size_t)(b * NN + mt * 32 + r) * DD + h * dh + c * 8);
      for (int j = 0; j < 8; ++j) Vt[c * 8 + j][r] = tv.h[j];
    }
    wait_async0();
    __syncthreads();

    v8f s0 = vzero, s1 = vzero;
    const int key = (lane & 15);
    for (int c = 0; c < 4; ++c) {
      Frag bk0, bk1;
      for (int v = 0; v < 8; ++v) {
        int k = c * 32 + b_kidx(v, lane);
        bk0.u[v] = *reinterpret_cast<const unsigned int*>(&Ks[key][k]);
        bk1.u[v] = *reinterpret_cast<const unsigned int*>(&Ks[key + 16][k]);
      }
      s0 = wmma_bf16(aq[c].v, bk0.v, s0);
      s1 = wmma_bf16(aq[c].v, bk1.v, s1);
    }

    float alpha[8];
    for (int r = 0; r < 8; ++r) {
      float x0 = s0[r] * scale, x1 = s1[r] * scale;
      float mx = fmaxf(x0, x1);
      mx = fmaxf(mx, __shfl_xor(mx, 1));
      mx = fmaxf(mx, __shfl_xor(mx, 2));
      mx = fmaxf(mx, __shfl_xor(mx, 4));
      mx = fmaxf(mx, __shfl_xor(mx, 8));
      float mnew = fmaxf(mrow[r], mx);
      float a  = __expf(mrow[r] - mnew);
      float p0 = __expf(x0 - mnew);
      float p1 = __expf(x1 - mnew);
      float sum = p0 + p1;
      sum += __shfl_xor(sum, 1);
      sum += __shfl_xor(sum, 2);
      sum += __shfl_xor(sum, 4);
      sum += __shfl_xor(sum, 8);
      lrow[r]  = lrow[r] * a + sum;
      mrow[r]  = mnew;
      alpha[r] = a;
      s0[r] = p0; s1[r] = p1;
    }
    for (int nt = 0; nt < 8; ++nt)
      for (int r = 0; r < 8; ++r)
        acc[nt][r] *= alpha[r];

    {
      int col = (lane & 15);
      int rhi = (lane & 16) ? 8 : 0;
      for (int r = 0; r < 8; ++r) {
        Pb[wv][r + rhi][col]      = (__bf16)s0[r];
        Pb[wv][r + rhi][col + 16] = (__bf16)s1[r];
      }
    }
    asm volatile("s_wait_dscnt 0" ::: "memory");  // same-wave LDS is in-order
    Frag ap;
    {
      int prow = (lane & 15);
      for (int v = 0; v < 8; ++v)
        ap.u[v] = *reinterpret_cast<const unsigned int*>(&Pb[wv][prow][a_kidx(v, lane)]);
    }

    for (int nt = 0; nt < 8; ++nt) {
      Frag bvf;
      int col = nt * 16 + (lane & 15);
      for (int v = 0; v < 8; ++v)
        bvf.u[v] = *reinterpret_cast<const unsigned int*>(&Vt[col][b_kidx(v, lane)]);
      acc[nt] = wmma_bf16(ap.v, bvf.v, acc[nt]);
    }
    __syncthreads();
  }

  const int rbase = (lane & 16) ? 8 : 0;
  const int col   = (lane & 15);
  for (int r = 0; r < 8; ++r) {
    float inv = 1.0f / lrow[r];
    int nrow = blockIdx.y * 128 + wv * 16 + rbase + r;
    size_t base = (size_t)(b * NN + nrow) * DD + h * dh;
    for (int nt = 0; nt < 8; ++nt) {
      size_t o = base + nt * 16 + col;
      outp[o] = acc[nt][r] * inv + x[o];
    }
  }
}

extern "C" void kernel_launch(void* const* d_in, const int* in_sizes, int n_in,
                              void* d_out, int out_size, void* d_ws, size_t ws_size,
                              hipStream_t stream) {
  const float* x  = (const float*)d_in[0];
  const float* Wq = (const float*)d_in[1];
  const float* bq = (const float*)d_in[2];
  const float* Wk = (const float*)d_in[3];
  const float* bk = (const float*)d_in[4];
  const float* Wv = (const float*)d_in[5];
  const float* bv = (const float*)d_in[6];
  float* outp = (float*)d_out;

  const size_t XE = (size_t)BB * NN * DD;
  const size_t WE = (size_t)DD * DD;
  __bf16* xb  = (__bf16*)d_ws;
  __bf16* Wqt = xb  + XE;
  __bf16* Wkt = Wqt + WE;
  __bf16* Wvt = Wkt + WE;
  __bf16* Qb  = Wvt + WE;
  __bf16* Kb  = Qb  + XE;
  __bf16* Vb  = Kb  + XE;

  convert_bf16_kernel<<<4096, 256, 0, stream>>>(x, xb, (int)(XE / 4));
  dim3 gt(DD / 64, DD / 64);
  transpose_bf16_kernel<<<gt, 256, 0, stream>>>(Wq, Wqt);
  transpose_bf16_kernel<<<gt, 256, 0, stream>>>(Wk, Wkt);
  transpose_bf16_kernel<<<gt, 256, 0, stream>>>(Wv, Wvt);

  dim3 g1(DD / 128, (BB * NN) / 128, 3);
  qkv_gemm_kernel<<<g1, 256, 0, stream>>>(xb, Wqt, bq, Wkt, bk, Wvt, bv, Qb, Kb, Vb);

  dim3 g2(BB * HH, NN / 128);
  flash_attn_kernel<<<g2, 256, 0, stream>>>(x, Qb, Kb, Vb, outp);
}
